// TransformerNO_71863392796730
// MI455X (gfx1250) — matrix operand order repaired and checked by reading, not compiled, not run on hardware
//
#include <hip/hip_runtime.h>
#include <math.h>

// ---------------------------------------------------------------------------
// Types / helpers for CDNA5 WMMA (wave32)
// ---------------------------------------------------------------------------
typedef __attribute__((ext_vector_type(16))) __bf16 v16bf;
typedef __attribute__((ext_vector_type(8)))  float  v8f;
typedef unsigned short u16;

union Frag16 { v16bf v; unsigned u[8]; };

__device__ __forceinline__ u16 f2bf(float f) {
  union { float f; unsigned u; } x; x.f = f;
  unsigned r = x.u + 0x7FFFu + ((x.u >> 16) & 1u);   // round-to-nearest-even
  return (u16)(r >> 16);
}
__device__ __forceinline__ float bf2f(u16 h) {
  union { unsigned u; float f; } x; x.u = ((unsigned)h) << 16;
  return x.f;
}
__device__ __forceinline__ float gelu_f(float x) {
  return 0.5f * x * (1.0f + erff(x * 0.70710678118654752440f));
}

// ---------------------------------------------------------------------------
// WMMA GEMM:  C[z] = act( A[z] @ B[z]^T' + bias ) (+ res)     (all bf16 I/O)
//   A : bf16 [M,K]   row-major, row stride lda
//   B : bf16 [N,K]   row-major (pre-transposed weights), row stride ldb
//   C : bf16 [M,N]   row-major, row stride ldc; bias fp32
// Block tile 128x64, 256 threads = 8 waves; each wave -> 32x32 (4 WMMA/K-step).
// Inner loop is pure b128 copies + ds_load_b128 fragments + v_wmma (no cvt ALU).
// Batched over blockIdx.z with (b,h) = (z/H, z%H) pointer offsets.
// act: 0 = none, 1 = gelu, 2 = sin(scale*x)
// ---------------------------------------------------------------------------
struct GemmP {
  const u16* A; const u16* B; const float* bias; const u16* res; u16* C;
  int lda, ldb, ldc;
  int M, N, K, H;
  long long aB, aH, wB, wH, cB, cH;
  int act; float scale;
};

__global__ __launch_bounds__(256, 2) void gemm_kernel(GemmP p) {
  __shared__ u16 sA[128 * 32];   // A tile [m][k]
  __shared__ u16 sB[64 * 32];    // B tile [n][k]
  const int tid = threadIdx.x;
  const int z = blockIdx.z;
  const int bI = z / p.H, hI = z % p.H;
  const u16* A = p.A + (long long)bI * p.aB + (long long)hI * p.aH;
  const u16* B = p.B + (long long)bI * p.wB + (long long)hI * p.wH;
  u16*       C = p.C + (long long)bI * p.cB + (long long)hI * p.cH;
  const u16* R = p.res ? (p.res + (long long)bI * p.cB + (long long)hI * p.cH) : nullptr;

  const int m0 = blockIdx.y * 128, n0 = blockIdx.x * 64;
  const int wave = tid >> 5, lane = tid & 31;
  const int wm = wave >> 1, wn = wave & 1;     // 4 row-groups x 2 col-groups
  const int mh = lane & 15, hf = lane >> 4;    // fragment lane decomposition
  const int lr = tid >> 2, lk = (tid & 3) * 8; // tile loader (16B segments)

  v8f acc00 = {0.f,0.f,0.f,0.f,0.f,0.f,0.f,0.f};
  v8f acc01 = acc00, acc10 = acc00, acc11 = acc00;

  for (int kb = 0; kb < p.K; kb += 32) {
    const u16* ga0 = A + (long long)(m0 + lr) * p.lda + kb + lk;
    const u16* ga1 = ga0 + (long long)64 * p.lda;
    const u16* gb  = B + (long long)(n0 + lr) * p.ldb + kb + lk;
    *(uint4*)(sA + lr * 32 + lk)        = *(const uint4*)ga0;
    *(uint4*)(sA + (lr + 64) * 32 + lk) = *(const uint4*)ga1;
    *(uint4*)(sB + lr * 32 + lk)        = *(const uint4*)gb;
    if (kb + 32 < p.K) { __builtin_prefetch(ga0 + 32, 0, 0); __builtin_prefetch(ga1 + 32, 0, 0); }
    __syncthreads();

    // A fragments (16x32 bf16, ISA lane-half K interleave): rows wm*32+{0,16}
    Frag16 a0, a1, b0, b1;
    const u16* pa0 = sA + (wm * 32 + mh) * 32;
    const u16* pa1 = pa0 + 16 * 32;
    #pragma unroll
    for (int v = 0; v < 8; ++v) {
      int k = (v < 4) ? (8 * hf + 2 * v) : (16 + 8 * hf + 2 * (v - 4));
      a0.u[v] = *(const unsigned*)(pa0 + k);
      a1.u[v] = *(const unsigned*)(pa1 + k);
    }
    // B fragments (32x16): same lane-half K interleave as A
    const u16* pb0 = sB + (wn * 32 + mh) * 32;
    const u16* pb1 = pb0 + 16 * 32;
    #pragma unroll
    for (int v = 0; v < 8; ++v) {
      int k = (v < 4) ? (8 * hf + 2 * v) : (16 + 8 * hf + 2 * (v - 4));
      b0.u[v] = *(const unsigned*)(pb0 + k);
      b1.u[v] = *(const unsigned*)(pb1 + k);
    }
    acc00 = __builtin_amdgcn_wmma_f32_16x16x32_bf16(false, a0.v, false, b0.v, (short)0, acc00, false, false);
    acc01 = __builtin_amdgcn_wmma_f32_16x16x32_bf16(false, a0.v, false, b1.v, (short)0, acc01, false, false);
    acc10 = __builtin_amdgcn_wmma_f32_16x16x32_bf16(false, a1.v, false, b0.v, (short)0, acc10, false, false);
    acc11 = __builtin_amdgcn_wmma_f32_16x16x32_bf16(false, a1.v, false, b1.v, (short)0, acc11, false, false);
    __syncthreads();
  }

  // Epilogue: C/D layout -> VGPR v holds row (v + 8*half), col = lane&15
  const int c0 = n0 + wn * 32 + mh;
  const int r0 = m0 + wm * 32;
  auto emit = [&](v8f acc, int rbase, int cc) {
    #pragma unroll
    for (int v = 0; v < 8; ++v) {
      int r = rbase + v + 8 * hf;
      float x = acc[v];
      if (p.bias) x += p.bias[cc];
      if (p.act == 1)      x = gelu_f(x);
      else if (p.act == 2) x = sinf(p.scale * x);
      long long o = (long long)r * p.ldc + cc;
      if (R) x += bf2f(R[o]);
      C[o] = f2bf(x);
    }
  };
  emit(acc00, r0,      c0);
  emit(acc01, r0,      c0 + 16);
  emit(acc10, r0 + 16, c0);
  emit(acc11, r0 + 16, c0 + 16);
}

// ---------------------------------------------------------------------------
// K^T V reduction (linear-attention "dots"): per (b,h), dots[64,64] +=
//   (1/N) * sum_n K[n,d] * V[n,e].  bf16 inputs, fp32 atomic accumulation.
// ---------------------------------------------------------------------------
__global__ __launch_bounds__(256, 2) void ktv_kernel(
    const u16* __restrict__ Kp, const u16* __restrict__ Vp,
    float* __restrict__ dots, int N, int CH, int H, int nsplit, float invN) {
  __shared__ u16 sKT[64 * 32];  // [d][n]   (A = K^T)
  __shared__ u16 sVT[64 * 32];  // [e][n]   (B = V transposed in LDS)
  const int tid = threadIdx.x;
  const int z = blockIdx.y;
  const int b = z / H, h = z % H;
  const u16* Kb = Kp + (long long)b * N * CH + h * 64;
  const u16* Vb = Vp + (long long)b * N * CH + h * 64;
  const int chunk = N / nsplit;
  const int nbeg = blockIdx.x * chunk;

  const int wave = tid >> 5, lane = tid & 31;
  const int wm = wave >> 1, wn = wave & 1;
  const int mh = lane & 15, hf = lane >> 4;
  const int ld_d = tid & 63, ld_g = tid >> 6;

  v8f acc0 = {0.f,0.f,0.f,0.f,0.f,0.f,0.f,0.f};
  v8f acc1 = acc0;

  for (int nb = nbeg; nb < nbeg + chunk; nb += 32) {
    #pragma unroll
    for (int i = 0; i < 8; ++i) {
      int nl = ld_g * 8 + i;
      sKT[ld_d * 32 + nl] = Kb[(long long)(nb + nl) * CH + ld_d];
      sVT[ld_d * 32 + nl] = Vb[(long long)(nb + nl) * CH + ld_d];
    }
    __syncthreads();

    Frag16 a, b0, b1;
    const u16* par = sKT + (wm * 16 + mh) * 32;
    #pragma unroll
    for (int v = 0; v < 8; ++v) {
      int k = (v < 4) ? (8 * hf + 2 * v) : (16 + 8 * hf + 2 * (v - 4));
      a.u[v] = *(const unsigned*)(par + k);
    }
    const u16* pb0 = sVT + (wn * 32 + mh) * 32;
    const u16* pb1 = pb0 + 16 * 32;
    #pragma unroll
    for (int v = 0; v < 8; ++v) {
      int k = (v < 4) ? (8 * hf + 2 * v) : (16 + 8 * hf + 2 * (v - 4));
      b0.u[v] = *(const unsigned*)(pb0 + k);
      b1.u[v] = *(const unsigned*)(pb1 + k);
    }
    acc0 = __builtin_amdgcn_wmma_f32_16x16x32_bf16(false, a.v, false, b0.v, (short)0, acc0, false, false);
    acc1 = __builtin_amdgcn_wmma_f32_16x16x32_bf16(false, a.v, false, b1.v, (short)0, acc1, false, false);
    __syncthreads();
  }

  const int c0 = wn * 32 + mh;
  float* dz = dots + (long long)z * 4096;
  #pragma unroll
  for (int v = 0; v < 8; ++v) {
    int r = wm * 16 + v + 8 * hf;
    atomicAdd(&dz[r * 64 + c0],      acc0[v] * invN);
    atomicAdd(&dz[r * 64 + c0 + 16], acc1[v] * invN);
  }
}

// ---------------------------------------------------------------------------
// Weight pre-pass: fp32 [K,N] -> bf16 transposed [N,K]
// ---------------------------------------------------------------------------
__global__ __launch_bounds__(256) void wtr_kernel(
    const float* __restrict__ W, u16* __restrict__ Wt, int K, int N) {
  int i = blockIdx.x * 256 + threadIdx.x;
  if (i >= K * N) return;
  int n = i / K, k = i - n * K;
  Wt[i] = f2bf(W[k * N + n]);
}

// dots fp32 [z][d][e] -> bf16 transposed [z][e][d] (B-operand for Q @ dots)
__global__ __launch_bounds__(256) void dtr_kernel(
    const float* __restrict__ D, u16* __restrict__ Dt, int Z) {
  int i = blockIdx.x * 256 + threadIdx.x;
  if (i >= Z * 4096) return;
  int z = i >> 12, r = (i >> 6) & 63, c = i & 63;
  Dt[i] = f2bf(D[(z << 12) + c * 64 + r]);
}

// ---------------------------------------------------------------------------
// LayerNorm over C=256 (bf16 in/out), one wave per row, shfl reductions.
// ---------------------------------------------------------------------------
__global__ __launch_bounds__(256) void ln_kernel(
    const u16* __restrict__ x, const float* __restrict__ g,
    const float* __restrict__ b, u16* __restrict__ y, long long M) {
  long long row = (long long)blockIdx.x * 8 + (threadIdx.x >> 5);
  int lane = threadIdx.x & 31;
  const u16* xr = x + row * 256;
  float v[8]; float s = 0.f;
  #pragma unroll
  for (int j = 0; j < 8; ++j) { v[j] = bf2f(xr[lane + 32 * j]); s += v[j]; }
  #pragma unroll
  for (int o = 16; o > 0; o >>= 1) s += __shfl_xor(s, o, 32);
  float mean = s * (1.f / 256.f);
  float q = 0.f;
  #pragma unroll
  for (int j = 0; j < 8; ++j) { float d = v[j] - mean; q += d * d; }
  #pragma unroll
  for (int o = 16; o > 0; o >>= 1) q += __shfl_xor(q, o, 32);
  float rstd = rsqrtf(q * (1.f / 256.f) + 1e-5f);
  u16* yr = y + row * 256;
  #pragma unroll
  for (int j = 0; j < 8; ++j) {
    int c = lane + 32 * j;
    yr[c] = f2bf((v[j] - mean) * rstd * g[c] + b[c]);
  }
}

// ---------------------------------------------------------------------------
// Instance norm over sequence axis (per batch, per channel), bf16 in place.
// grid = (B, CH/64); block = 256 (64 channels x 4 row-groups).
// ---------------------------------------------------------------------------
__global__ __launch_bounds__(256) void instnorm_kernel(
    u16* __restrict__ x, int N, int CH) {
  const int b = blockIdx.x;
  const int c0 = blockIdx.y * 64;
  const int c = threadIdx.x & 63, rg = threadIdx.x >> 6;
  __shared__ float ss[4][64], sq[4][64], smean[64], srstd[64];
  float s = 0.f, q = 0.f;
  u16* xb = x + (long long)b * N * CH + c0 + c;
  for (int n = rg; n < N; n += 4) { float v = bf2f(xb[(long long)n * CH]); s += v; q += v * v; }
  ss[rg][c] = s; sq[rg][c] = q;
  __syncthreads();
  if (threadIdx.x < 64) {
    int t = threadIdx.x;
    float S = ss[0][t] + ss[1][t] + ss[2][t] + ss[3][t];
    float Q = sq[0][t] + sq[1][t] + sq[2][t] + sq[3][t];
    float m = S / (float)N;
    smean[t] = m;
    srstd[t] = rsqrtf(Q / (float)N - m * m + 1e-5f);
  }
  __syncthreads();
  float m = smean[c], r = srstd[c];
  for (int n = rg; n < N; n += 4) {
    long long o = (long long)n * CH;
    xb[o] = f2bf((bf2f(xb[o]) - m) * r);
  }
}

// ---------------------------------------------------------------------------
// 2-D rotary, bf16 in place. Per head of 64: dims [0,32) use pos.x freqs,
// dims [32,64) use pos.y; rotation pairs (j, j+16) inside each 32-half.
// ---------------------------------------------------------------------------
__global__ __launch_bounds__(256) void rotary_kernel(
    u16* __restrict__ t, const float* __restrict__ pos, long long M, int CH) {
  long long i = (long long)blockIdx.x * 256 + threadIdx.x;
  long long total = M * (long long)(CH >> 1);
  if (i >= total) return;
  long long m = i / (CH >> 1);
  int pc = (int)(i - m * (CH >> 1));
  int h = pc >> 5;
  int p = pc & 31;
  int j = p & 15;
  int sel = p >> 4;                        // 0 -> x coord, 1 -> y coord
  float coord = pos[m * 2 + sel] * 64.0f;  // / MIN_FREQ
  float f = coord * __expf(-(float)j * 0.57564627324851145f);  // ln(1e4)/16
  float sf, cf; __sincosf(f, &sf, &cf);
  long long i0 = m * CH + h * 64 + sel * 32 + j;
  long long i1 = i0 + 16;
  float x0 = bf2f(t[i0]), x1 = bf2f(t[i1]);
  t[i0] = f2bf(x0 * cf - x1 * sf);
  t[i1] = f2bf(x1 * cf + x0 * sf);
}

// Lift: fp32 [M,3] @ fp32 [3,256] + b -> bf16
__global__ __launch_bounds__(256) void lift_kernel(
    const float* __restrict__ u, const float* __restrict__ W,
    const float* __restrict__ b, u16* __restrict__ out, long long M) {
  long long i = (long long)blockIdx.x * 256 + threadIdx.x;
  if (i >= M * 256) return;
  long long m = i >> 8; int c = (int)(i & 255);
  out[i] = f2bf(u[m * 3] * W[c] + u[m * 3 + 1] * W[256 + c] + u[m * 3 + 2] * W[512 + c] + b[c]);
}

// SIREN first layer: sin(30 * ([M,2] @ [2,256] + b)) -> bf16
__global__ __launch_bounds__(256) void siren_first_kernel(
    const float* __restrict__ pos, const float* __restrict__ W,
    const float* __restrict__ b, u16* __restrict__ out, long long M) {
  long long i = (long long)blockIdx.x * 256 + threadIdx.x;
  if (i >= M * 256) return;
  long long m = i >> 8; int c = (int)(i & 255);
  out[i] = f2bf(sinf(30.f * (pos[m * 2] * W[c] + pos[m * 2 + 1] * W[256 + c] + b[c])));
}

// Final projection: bf16 [M,256] @ fp32 [256,1] + b -> fp32; one wave per row.
__global__ __launch_bounds__(256) void proj_out_kernel(
    const u16* __restrict__ x, const float* __restrict__ W,
    const float* __restrict__ b, float* __restrict__ out, long long M) {
  long long row = (long long)blockIdx.x * 8 + (threadIdx.x >> 5);
  int lane = threadIdx.x & 31;
  const u16* xr = x + row * 256;
  float s = 0.f;
  #pragma unroll
  for (int j = 0; j < 8; ++j) { int c = lane + 32 * j; s += bf2f(xr[c]) * W[c]; }
  #pragma unroll
  for (int o = 16; o > 0; o >>= 1) s += __shfl_xor(s, o, 32);
  if (lane == 0) out[row] = s + b[0];
}

// ---------------------------------------------------------------------------
// Host-side orchestration
// ---------------------------------------------------------------------------
static void launch_gemm(hipStream_t s, const u16* A, int lda, const u16* B, int ldb,
                        const float* bias, const u16* res, u16* C, int ldc,
                        int M, int N, int K, int Z, int H,
                        long long aB, long long aH, long long wB, long long wH,
                        long long cB, long long cH, int act, float scale) {
  GemmP p{A, B, bias, res, C, lda, ldb, ldc, M, N, K, H, aB, aH, wB, wH, cB, cH, act, scale};
  dim3 g((unsigned)(N / 64), (unsigned)(M / 128), (unsigned)Z);
  hipLaunchKernelGGL(gemm_kernel, g, dim3(256), 0, s, p);
}

static void gemm_plain(hipStream_t s, const u16* A, const u16* Bt,
                       const float* bias, const u16* res, u16* C,
                       int M, int N, int K, int act, float scale) {
  launch_gemm(s, A, K, Bt, K, bias, res, C, N, M, N, K, 1, 1, 0, 0, 0, 0, 0, 0, act, scale);
}

extern "C" void kernel_launch(void* const* d_in, const int* in_sizes, int n_in,
                              void* d_out, int out_size, void* d_ws, size_t ws_size,
                              hipStream_t stream) {
  (void)in_sizes; (void)n_in; (void)out_size;
  const long long Mt = 65536;                 // B * N tokens
  const int Nseq = 16384, Bb = 4;
  const float invN = 1.0f / 16384.0f;

  auto F = [&](int i) { return (const float*)d_in[i]; };
  // JAX pytree (sorted dict keys) flat order:
  const float* dec_Wk = F(0);  const float* dec_Wo = F(1);
  const float* dec_Wv = F(2);  const float* dec_bo = F(3);
  // enc layers at 4 + 13*l: [W1,W2,Wk,Wo,Wq,Wv,b1,b2,bo,ln1_b,ln1_g,ln2_b,ln2_g]
  const float* lift_W = F(43); const float* lift_b = F(44);
  const float* pn_b = F(45);   const float* pn_g = F(46);
  const float* proj_W1 = F(47); const float* proj_W2 = F(48);
  const float* proj_b1 = F(49); const float* proj_b2 = F(50);
  // siren: 51..58 (W0,b0,W1,b1,W2,b2,W3,b3); 59 siren_Wl; 60 siren_bl
  const float* pos_qry = F(61);
  const float* pos_src = F(62);
  const float* u_in    = F(63);

  if (ws_size < ((size_t)400 << 20)) return;
  char* wp = (char*)d_ws;
  auto carveB = [&](size_t bytes) { void* r = (void*)wp; wp += (bytes + 255) & ~(size_t)255; return r; };
  // bf16 activation buffers
  u16* BU = (u16*)carveB((size_t)Mt * 256 * 2);   // residual stream u
  u16* BX = (u16*)carveB((size_t)Mt * 256 * 2);   // LN output / scratch
  u16* BQ = (u16*)carveB((size_t)Mt * 512 * 2);   // q / query_emb
  u16* BK = (u16*)carveB((size_t)Mt * 512 * 2);   // k
  u16* BV = (u16*)carveB((size_t)Mt * 512 * 2);   // v
  u16* BA = (u16*)carveB((size_t)Mt * 512 * 2);   // attn-merged / MLP hidden
  float* BD = (float*)carveB((size_t)32 * 4096 * 4);  // dots fp32
  u16*  BDt = (u16*)carveB((size_t)32 * 4096 * 2);    // dots bf16 transposed
  // bf16 transposed weights
  u16 *tWq[3], *tWk[3], *tWv[3], *tWo[3], *tW1[3], *tW2[3];
  for (int l = 0; l < 3; ++l) {
    tWq[l] = (u16*)carveB(65536 * 2);  tWk[l] = (u16*)carveB(65536 * 2);
    tWv[l] = (u16*)carveB(65536 * 2);  tWo[l] = (u16*)carveB(65536 * 2);
    tW1[l] = (u16*)carveB(131072 * 2); tW2[l] = (u16*)carveB(131072 * 2);
  }
  u16* tS1 = (u16*)carveB(65536 * 2);
  u16* tS2 = (u16*)carveB(65536 * 2);
  u16* tS3 = (u16*)carveB(65536 * 2);
  u16* tSl = (u16*)carveB(131072 * 2);
  u16* tDk = (u16*)carveB(131072 * 2);
  u16* tDv = (u16*)carveB(131072 * 2);
  u16* tDo = (u16*)carveB(131072 * 2);
  u16* tP1 = (u16*)carveB(65536 * 2);

  auto wtr = [&](const float* W, u16* Wt, int K, int N) {
    int tot = K * N;
    hipLaunchKernelGGL(wtr_kernel, dim3((unsigned)((tot + 255) / 256)), dim3(256), 0, stream, W, Wt, K, N);
  };
  // ---- weight pre-pass: fp32 [K,N] -> bf16 [N,K] ----
  for (int l = 0; l < 3; ++l) {
    int e = 4 + 13 * l;
    wtr(F(e + 4), tWq[l], 256, 256); wtr(F(e + 2), tWk[l], 256, 256);
    wtr(F(e + 5), tWv[l], 256, 256); wtr(F(e + 3), tWo[l], 256, 256);
    wtr(F(e + 0), tW1[l], 256, 512); wtr(F(e + 1), tW2[l], 512, 256);
  }
  wtr(F(53), tS1, 256, 256); wtr(F(55), tS2, 256, 256); wtr(F(57), tS3, 256, 256);
  wtr(F(59), tSl, 256, 512);
  wtr(dec_Wk, tDk, 256, 512); wtr(dec_Wv, tDv, 256, 512); wtr(dec_Wo, tDo, 512, 256);
  wtr(proj_W1, tP1, 256, 256);

  const unsigned ew = (unsigned)((Mt * 256) / 256);   // elementwise grid
  const unsigned lnb = (unsigned)(Mt / 8);

  // ---- lift ----
  hipLaunchKernelGGL(lift_kernel, dim3(ew), dim3(256), 0, stream, u_in, lift_W, lift_b, BU, Mt);

  // ---- encoder layers ----
  for (int l = 0; l < 3; ++l) {
    int e = 4 + 13 * l;
    const float* b1   = F(e + 6);  const float* b2   = F(e + 7);
    const float* bo   = F(e + 8);
    const float* ln1b = F(e + 9);  const float* ln1g = F(e + 10);
    const float* ln2b = F(e + 11); const float* ln2g = F(e + 12);

    hipLaunchKernelGGL(ln_kernel, dim3(lnb), dim3(256), 0, stream, BU, ln1g, ln1b, BX, Mt);
    gemm_plain(stream, BX, tWq[l], nullptr, nullptr, BQ, (int)Mt, 256, 256, 0, 0.f);
    gemm_plain(stream, BX, tWk[l], nullptr, nullptr, BK, (int)Mt, 256, 256, 0, 0.f);
    gemm_plain(stream, BX, tWv[l], nullptr, nullptr, BV, (int)Mt, 256, 256, 0, 0.f);
    hipLaunchKernelGGL(instnorm_kernel, dim3(Bb, 4), dim3(256), 0, stream, BK, Nseq, 256);
    hipLaunchKernelGGL(instnorm_kernel, dim3(Bb, 4), dim3(256), 0, stream, BV, Nseq, 256);
    {
      unsigned rb = (unsigned)((Mt * 128 + 255) / 256);
      hipLaunchKernelGGL(rotary_kernel, dim3(rb), dim3(256), 0, stream, BQ, pos_src, Mt, 256);
      hipLaunchKernelGGL(rotary_kernel, dim3(rb), dim3(256), 0, stream, BK, pos_src, Mt, 256);
    }
    hipMemsetAsync(BD, 0, (size_t)16 * 4096 * 4, stream);
    hipLaunchKernelGGL(ktv_kernel, dim3(32, 16), dim3(256), 0, stream,
                       BK, BV, BD, Nseq, 256, 4, 32, invN);
    hipLaunchKernelGGL(dtr_kernel, dim3((unsigned)((16 * 4096 + 255) / 256)), dim3(256), 0, stream,
                       BD, BDt, 16);
    // per-(b,h): out = Q @ dots -> merged heads in BA
    launch_gemm(stream, BQ, 256, BDt, 64, nullptr, nullptr, BA, 256,
                Nseq, 64, 64, 16, 4,
                (long long)Nseq * 256, 64, (long long)4 * 4096, 4096,
                (long long)Nseq * 256, 64, 0, 0.f);
    // u = u + BA @ Wo + bo
    gemm_plain(stream, BA, tWo[l], bo, BU, BU, (int)Mt, 256, 256, 0, 0.f);
    // MLP
    hipLaunchKernelGGL(ln_kernel, dim3(lnb), dim3(256), 0, stream, BU, ln2g, ln2b, BX, Mt);
    gemm_plain(stream, BX, tW1[l], b1, nullptr, BA, (int)Mt, 512, 256, 1, 0.f);  // gelu
    gemm_plain(stream, BA, tW2[l], b2, BU, BU, (int)Mt, 256, 512, 0, 0.f);       // + residual
  }

  // ---- SIREN query network ----
  hipLaunchKernelGGL(siren_first_kernel, dim3(ew), dim3(256), 0, stream,
                     pos_qry, F(51), F(52), BX, Mt);
  gemm_plain(stream, BX, tS1, F(54), nullptr, BA, (int)Mt, 256, 256, 2, 1.0f);
  gemm_plain(stream, BA, tS2, F(56), nullptr, BX, (int)Mt, 256, 256, 2, 1.0f);
  gemm_plain(stream, BX, tS3, F(58), nullptr, BA, (int)Mt, 256, 256, 2, 1.0f);
  gemm_plain(stream, BA, tSl, F(60), nullptr, BQ, (int)Mt, 512, 256, 0, 0.f);  // query_emb

  // ---- decoder cross attention (H=8, D=64) ----
  gemm_plain(stream, BU, tDk, nullptr, nullptr, BK, (int)Mt, 512, 256, 0, 0.f);
  gemm_plain(stream, BU, tDv, nullptr, nullptr, BV, (int)Mt, 512, 256, 0, 0.f);
  hipLaunchKernelGGL(instnorm_kernel, dim3(Bb, 8), dim3(256), 0, stream, BK, Nseq, 512);
  hipLaunchKernelGGL(instnorm_kernel, dim3(Bb, 8), dim3(256), 0, stream, BV, Nseq, 512);
  {
    unsigned rb = (unsigned)((Mt * 256 + 255) / 256);
    hipLaunchKernelGGL(rotary_kernel, dim3(rb), dim3(256), 0, stream, BQ, pos_qry, Mt, 512);
    hipLaunchKernelGGL(rotary_kernel, dim3(rb), dim3(256), 0, stream, BK, pos_src, Mt, 512);
  }
  hipMemsetAsync(BD, 0, (size_t)32 * 4096 * 4, stream);
  hipLaunchKernelGGL(ktv_kernel, dim3(32, 32), dim3(256), 0, stream,
                     BK, BV, BD, Nseq, 512, 8, 32, invN);
  hipLaunchKernelGGL(dtr_kernel, dim3((unsigned)((32 * 4096 + 255) / 256)), dim3(256), 0, stream,
                     BD, BDt, 32);
  launch_gemm(stream, BQ, 512, BDt, 64, nullptr, nullptr, BA, 512,
              Nseq, 64, 64, 32, 8,
              (long long)Nseq * 512, 64, (long long)8 * 4096, 4096,
              (long long)Nseq * 512, 64, 0, 0.f);
  // u_out = BA @ dec_Wo + dec_bo
  gemm_plain(stream, BA, tDo, dec_bo, nullptr, BX, (int)Mt, 256, 512, 0, 0.f);

  // ---- output head ----
  hipLaunchKernelGGL(ln_kernel, dim3(lnb), dim3(256), 0, stream, BX, pn_g, pn_b, BU, Mt);
  gemm_plain(stream, BU, tP1, proj_b1, nullptr, BA, (int)Mt, 256, 256, 1, 0.f); // gelu
  hipLaunchKernelGGL(proj_out_kernel, dim3(lnb), dim3(256), 0, stream,
                     BA, proj_W2, proj_b2, (float*)d_out, Mt);
}
